// BINLayer_36867999269443
// MI455X (gfx1250) — 8-bit matrix operand order repaired and checked by reading, compiled, not run on hardware
//
#include <hip/hip_runtime.h>
#include <cstdint>

typedef __attribute__((ext_vector_type(8))) int v8i;

#define D 128
#define BM 128
#define THREADS 256
#define ROW_U32 34

__device__ __forceinline__ int sgn8(float x) {
    return (x > 0.0f) ? 1 : ((x < 0.0f) ? -1 : 0);
}

__device__ __forceinline__ uint32_t pack4(float a, float b, float c, float d) {
    uint32_t r =  (uint32_t)(uint8_t)(int8_t)sgn8(a);
    r |= ((uint32_t)(uint8_t)(int8_t)sgn8(b)) << 8;
    r |= ((uint32_t)(uint8_t)(int8_t)sgn8(c)) << 16;
    r |= ((uint32_t)(uint8_t)(int8_t)sgn8(d)) << 24;
    return r;
}

__device__ __forceinline__ float fast_tanh(float x) {
    float t = __builtin_amdgcn_exp2f(x * 2.885390081777926815f);
    float r = __builtin_amdgcn_rcpf(t + 1.0f);
    return __builtin_fmaf(-2.0f, r, 1.0f);
}

__global__ void pack_kernel_sgn(const float* __restrict__ K, uint32_t* __restrict__ wsB) {
    int tid = blockIdx.x * blockDim.x + threadIdx.x;
    if (tid >= 8 * 2 * 32 * 8) return;
    int j    = tid & 7;
    int lane = (tid >> 3) & 31;
    int s    = (tid >> 8) & 1;
    int t    = tid >> 9;
    int n    = 16 * t + (lane & 15);
    int kb   = 64 * s + 16 * (j >> 1) + 8 * (lane >> 4) + 4 * (j & 1);
    float a = K[(size_t)(kb + 0) * D + n];
    float b = K[(size_t)(kb + 1) * D + n];
    float c = K[(size_t)(kb + 2) * D + n];
    float d = K[(size_t)(kb + 3) * D + n];
    wsB[tid] = pack4(a, b, c, d);
}

__global__ void __launch_bounds__(THREADS, 2)
bin_gemm_tanh(const float* __restrict__ in, const uint32_t* __restrict__ wsB,
              const float* __restrict__ bias, float* __restrict__ out) {
    __shared__ uint32_t smem[BM * ROW_U32];

    const int tid = threadIdx.x;
    const size_t row0 = (size_t)blockIdx.x * BM;
    const float4* __restrict__ inv = (const float4*)(in + row0 * D);

    #pragma unroll
    for (int i = 0; i < 16; ++i) {
        int d = tid + THREADS * i;
        float4 v = inv[d];
        smem[(d >> 5) * ROW_U32 + (d & 31)] = pack4(v.x, v.y, v.z, v.w);
    }
    __syncthreads();

    const int wave = tid >> 5;
    const int lane = tid & 31;
    const int m0   = wave * 16;
    const int half = lane >> 4;
    const int mr   = lane & 15;

    v8i a0, a1;
    {
        const uint32_t* rowp = &smem[(m0 + mr) * ROW_U32];
        #pragma unroll
        for (int p = 0; p < 4; ++p) {
            uint2 q0 = *(const uint2*)&rowp[4 * p + 2 * half];
            a0[2 * p]     = (int)q0.x;
            a0[2 * p + 1] = (int)q0.y;
            uint2 q1 = *(const uint2*)&rowp[16 + 4 * p + 2 * half];
            a1[2 * p]     = (int)q1.x;
            a1[2 * p + 1] = (int)q1.y;
        }
    }

    const int mbase = m0 + 8 * half;
    float* __restrict__ outp = out + (row0 + mbase) * D + mr;

    #pragma unroll
    for (int t = 0; t < 8; ++t) {
        v8i b0 = *(const v8i*)(wsB + ((size_t)(t * 2 + 0) * 32 + lane) * 8);
        v8i b1 = *(const v8i*)(wsB + ((size_t)(t * 2 + 1) * 32 + lane) * 8);
        v8i acc = {};
        acc = __builtin_amdgcn_wmma_i32_16x16x64_iu8(true, a0, true, b0, acc, false, false);
        acc = __builtin_amdgcn_wmma_i32_16x16x64_iu8(true, a1, true, b1, acc, false, false);
        float bv = bias[16 * t + mr];
        #pragma unroll
        for (int e = 0; e < 8; ++e) {
            outp[(size_t)e * D + 16 * t] = fast_tanh((float)acc[e] + bv);
        }
    }
}

extern "C" void kernel_launch(void* const* d_in, const int* in_sizes, int n_in,
                              void* d_out, int out_size, void* d_ws, size_t ws_size,
                              hipStream_t stream) {
    const float* inputs = (const float*)d_in[0];
    const float* kern   = (const float*)d_in[1];
    const float* bias   = (const float*)d_in[2];
    float* outp = (float*)d_out;
    uint32_t* wsB = (uint32_t*)d_ws;

    int nrows = in_sizes[0] / D;
    pack_kernel_sgn<<<16, 256, 0, stream>>>(kern, wsB);
    bin_gemm_tanh<<<nrows / BM, THREADS, 0, stream>>>(inputs, wsB, bias, outp);
}
